// AttentionBlock_8985071583478
// MI455X (gfx1250) — matrix operand order repaired and checked by reading, not compiled, not run on hardware
//
#include <hip/hip_runtime.h>
#include <hip/hip_bf16.h>

typedef __attribute__((ext_vector_type(16))) _Float16 v16h;
typedef __attribute__((ext_vector_type(8)))  float    v8f;

union F16Frag { v16h v; _Float16 h[16]; };

#define DHEAD 256
#define NROWS 8192

// ---- WMMA fragment loaders (row-major source) -------------------------------
// A-matrix 16x32 f16 layout (ISA 7.12.2): lane m = lane&15, hi = lane>>4.
// halves 0..7  -> K = kOff + 8*hi + (0..7)
// halves 8..15 -> K = kOff + 16 + 8*hi + (0..7)   (gap of 16 halves is real)
__device__ __forceinline__ v16h load_frag_a(const _Float16* base, int ld, int kOff, int lane) {
    int m = lane & 15, hi = lane >> 4;
    const _Float16* p = base + m * ld + kOff + 8 * hi;
    F16Frag f;
    __builtin_memcpy(&f.h[0], p,      16);
    __builtin_memcpy(&f.h[8], p + 16, 16);
    return f.v;
}

// B-matrix 32x16 f16 layout: lane n = lane&15, halves 0..7 -> K = kOff + 8*hi + (0..7),
// halves 8..15 -> K = kOff + 16 + 8*hi + (0..7), in row-major W[n][k] (B = W^T).
__device__ __forceinline__ v16h load_frag_b(const _Float16* base, int ld, int kOff, int lane) {
    int n = lane & 15, hi = lane >> 4;
    const _Float16* p = base + n * ld + kOff + 8 * hi;
    F16Frag f;
    __builtin_memcpy(&f.h[0], p,      16);
    __builtin_memcpy(&f.h[8], p + 16, 16);
    return f.v;
}

// ---- gfx1250 async global->LDS (ASYNCcnt-tracked, no VGPR round trip) -------
// GVS form: vdst holds the LDS address, saddr + vaddr the global one; 16B per lane.
__device__ __forceinline__ void async_load_lds_b128(unsigned lds_off, const void* sbase,
                                                    unsigned voff) {
    asm volatile("global_load_async_to_lds_b128 %0, %1, %2"
                 :
                 : "v"(lds_off), "v"(voff), "s"(sbase)
                 : "memory");
}
__device__ __forceinline__ void wait_asynccnt0() {
    asm volatile("s_wait_asynccnt 0" ::: "memory");
}

// ---- Kernel 1: LayerNorm, fp32 -> f16 ---------------------------------------
__global__ void ln_kernel(const float* __restrict__ x, const float* __restrict__ w,
                          const float* __restrict__ b, _Float16* __restrict__ xn) {
    __shared__ float red[2][8];
    int row = blockIdx.x, tid = threadIdx.x;
    float val = x[(size_t)row * DHEAD + tid];
    float s = val, s2 = val * val;
    #pragma unroll
    for (int m = 16; m >= 1; m >>= 1) {
        s  += __shfl_xor(s,  m, 32);
        s2 += __shfl_xor(s2, m, 32);
    }
    int wave = tid >> 5, lane = tid & 31;
    if (lane == 0) { red[0][wave] = s; red[1][wave] = s2; }
    __syncthreads();
    float ts = 0.f, ts2 = 0.f;
    #pragma unroll
    for (int i = 0; i < 8; ++i) { ts += red[0][i]; ts2 += red[1][i]; }
    float mu  = ts  * (1.0f / DHEAD);
    float var = ts2 * (1.0f / DHEAD) - mu * mu;
    float inv = rsqrtf(var + 1e-5f);
    float o = (val - mu) * inv * w[tid] + b[tid];
    xn[(size_t)row * DHEAD + tid] = (_Float16)o;
}

// ---- Kernel 2: weight conversion fp32 -> f16 --------------------------------
__global__ void cvt_kernel(const float* __restrict__ qkvw, const float* __restrict__ projw,
                           _Float16* __restrict__ qkvw_h, _Float16* __restrict__ projw_h) {
    int i = blockIdx.x * blockDim.x + threadIdx.x;
    if (i < 768 * 256) qkvw_h[i] = (_Float16)qkvw[i];
    if (i < 256 * 256) projw_h[i] = (_Float16)projw[i];
}

// ---- Kernel 3: QKV GEMM  qkv = xn @ W^T + b, split into q/k/v (f16) ---------
__global__ void qkv_gemm(const _Float16* __restrict__ xn, const _Float16* __restrict__ wq,
                         const float* __restrict__ bias, _Float16* __restrict__ q,
                         _Float16* __restrict__ k, _Float16* __restrict__ v) {
    int wave = blockIdx.x * (blockDim.x >> 5) + (threadIdx.x >> 5);
    int lane = threadIdx.x & 31;
    int tM = wave / 48, tN = wave % 48;               // 512 x 48 tiles
    const _Float16* arow = xn + (size_t)tM * 16 * DHEAD;
    const _Float16* brow = wq + (size_t)tN * 16 * DHEAD;
    v8f acc = {};
    #pragma unroll
    for (int c = 0; c < 8; ++c) {
        v16h a = load_frag_a(arow, DHEAD, c * 32, lane);
        v16h b = load_frag_b(brow, DHEAD, c * 32, lane);
        acc = __builtin_amdgcn_wmma_f32_16x16x32_f16(false, a, false, b, (short)0, acc, false, false);
    }
    int lo = lane & 15, hi = lane >> 4;
    int n = tN * 16 + lo;
    float bn = bias[n];
    _Float16* dst = (n < 256) ? q : (n < 512 ? k : v);
    int colLocal = n & 255;
    #pragma unroll
    for (int r = 0; r < 8; ++r) {
        int m = tM * 16 + r + 8 * hi;
        dst[(size_t)m * DHEAD + colLocal] = (_Float16)(acc[r] + bn);
    }
}

// ---- Kernel 4: flash attention ----------------------------------------------
// 4 waves/block; wave owns 16 query rows, full 16x256 fp32 accumulator.
// K staged via async global->LDS (ASYNCcnt); V staged transposed through VGPRs.
__global__ __launch_bounds__(128, 1)
void flash_kernel(const _Float16* __restrict__ qg, const _Float16* __restrict__ kg,
                  const _Float16* __restrict__ vg, _Float16* __restrict__ og) {
    __shared__ _Float16 Ks[32 * 256];     // 16 KB, [row][col]
    __shared__ _Float16 Vt[256 * 32];     // 16 KB, [col][row]
    __shared__ _Float16 Ps[4][16 * 32];   //  4 KB, per-wave P tile

    int tid = threadIdx.x;
    int wave = tid >> 5, lane = tid & 31;
    int lo = lane & 15, hi = lane >> 4;
    int qRow = blockIdx.x * 64 + wave * 16;           // this wave's first query row

    // LDS byte offset of Ks (generic shared ptr low 32 bits == AS(3) offset)
    unsigned ksBase = (unsigned)(uintptr_t)(void*)&Ks[0];

    v8f acc[16];
    #pragma unroll
    for (int t = 0; t < 16; ++t) acc[t] = (v8f){};
    float rm[8], rl[8];
    #pragma unroll
    for (int r = 0; r < 8; ++r) { rm[r] = -1e30f; rl[r] = 0.f; }
    const float scale = 0.0625f;                      // 1/sqrt(256)
    const _Float16* qbase = qg + (size_t)qRow * DHEAD;

    for (int j = 0; j < NROWS; j += 32) {
        __syncthreads();
        // Stage K rows j..j+31 via async-to-LDS: 8192 halves, 16B per lane per op
        for (int i = tid * 8; i < 32 * 256; i += 128 * 8)
            async_load_lds_b128(ksBase + (unsigned)(i * 2), kg,
                                (unsigned)((j * 256 + i) * 2));
        // Stage V rows j..j+31 transposed: Vt[col][row]
        for (int i = tid * 8; i < 32 * 256; i += 128 * 8) {
            int row = i >> 8, col = i & 255;
            _Float16 tmp[8];
            __builtin_memcpy(tmp, vg + (size_t)(j + row) * DHEAD + col, 16);
            #pragma unroll
            for (int e = 0; e < 8; ++e) Vt[(col + e) * 32 + row] = tmp[e];
        }
        if (j + 32 < NROWS) {
            __builtin_prefetch(kg + (size_t)(j + 32) * DHEAD, 0, 1);
            __builtin_prefetch(vg + (size_t)(j + 32) * DHEAD, 0, 1);
        }
        wait_asynccnt0();
        __syncthreads();

        // S = q @ k^T for 16 rows x 32 keys (two 16x16 tiles)
        v8f s0 = {}, s1 = {};
        #pragma unroll
        for (int c = 0; c < 8; ++c) {
            v16h a  = load_frag_a(qbase, DHEAD, c * 32, lane);
            v16h b0 = load_frag_b(&Ks[0],        DHEAD, c * 32, lane);
            v16h b1 = load_frag_b(&Ks[16 * 256], DHEAD, c * 32, lane);
            s0 = __builtin_amdgcn_wmma_f32_16x16x32_f16(false, a, false, b0, (short)0, s0, false, false);
            s1 = __builtin_amdgcn_wmma_f32_16x16x32_f16(false, a, false, b1, (short)0, s1, false, false);
        }

        // Online softmax: rows live in 16-lane halves -> shfl_xor masks 1..8
        float al[8];
        #pragma unroll
        for (int r = 0; r < 8; ++r) {
            float v0 = s0[r] * scale, v1 = s1[r] * scale;
            float mx = fmaxf(v0, v1);
            #pragma unroll
            for (int m = 8; m >= 1; m >>= 1) mx = fmaxf(mx, __shfl_xor(mx, m, 32));
            float nm = fmaxf(rm[r], mx);
            al[r] = __expf(rm[r] - nm);
            float p0 = __expf(v0 - nm), p1 = __expf(v1 - nm);
            s0[r] = p0; s1[r] = p1;
            float sum = p0 + p1;
            #pragma unroll
            for (int m = 8; m >= 1; m >>= 1) sum += __shfl_xor(sum, m, 32);
            rl[r] = rl[r] * al[r] + sum;
            rm[r] = nm;
        }
        #pragma unroll
        for (int t = 0; t < 16; ++t)
            #pragma unroll
            for (int r = 0; r < 8; ++r) acc[t][r] *= al[r];

        // C-layout P -> LDS -> A-fragment
        _Float16* pp = &Ps[wave][0];
        #pragma unroll
        for (int r = 0; r < 8; ++r) {
            int row = r + 8 * hi;
            pp[row * 32 + lo]      = (_Float16)s0[r];
            pp[row * 32 + 16 + lo] = (_Float16)s1[r];
        }
        asm volatile("s_wait_dscnt 0" ::: "memory");   // wave-local LDS RAW fence
        v16h ap = load_frag_a(pp, 32, 0, lane);

        // O += P @ V  (16 column tiles, B-fragment contiguous in transposed Vt)
        #pragma unroll
        for (int t = 0; t < 16; ++t) {
            v16h bv = load_frag_b(&Vt[(size_t)t * 16 * 32], 32, 0, lane);
            acc[t] = __builtin_amdgcn_wmma_f32_16x16x32_f16(false, ap, false, bv, (short)0, acc[t], false, false);
        }
    }

    // Normalize and store (f16)
    #pragma unroll
    for (int r = 0; r < 8; ++r) rl[r] = 1.0f / rl[r];
    #pragma unroll
    for (int t = 0; t < 16; ++t)
        #pragma unroll
        for (int r = 0; r < 8; ++r) {
            int m = qRow + r + 8 * hi;
            int n = t * 16 + lo;
            og[(size_t)m * DHEAD + n] = (_Float16)(acc[t][r] * rl[r]);
        }
}

// ---- Kernel 5: proj GEMM + bias + residual, fp32 out ------------------------
__global__ void proj_gemm(const _Float16* __restrict__ ao, const _Float16* __restrict__ pw,
                          const float* __restrict__ pb, const float* __restrict__ x,
                          float* __restrict__ out) {
    int wave = blockIdx.x * (blockDim.x >> 5) + (threadIdx.x >> 5);
    int lane = threadIdx.x & 31;
    int tM = wave >> 4, tN = wave & 15;               // 512 x 16 tiles
    const _Float16* arow = ao + (size_t)tM * 16 * DHEAD;
    const _Float16* brow = pw + (size_t)tN * 16 * DHEAD;
    v8f acc = {};
    #pragma unroll
    for (int c = 0; c < 8; ++c) {
        v16h a = load_frag_a(arow, DHEAD, c * 32, lane);
        v16h b = load_frag_b(brow, DHEAD, c * 32, lane);
        acc = __builtin_amdgcn_wmma_f32_16x16x32_f16(false, a, false, b, (short)0, acc, false, false);
    }
    int lo = lane & 15, hi = lane >> 4;
    int n = tN * 16 + lo;
    float bn = pb[n];
    #pragma unroll
    for (int r = 0; r < 8; ++r) {
        int m = tM * 16 + r + 8 * hi;
        size_t idx = (size_t)m * DHEAD + n;
        out[idx] = x[idx] + acc[r] + bn;
    }
}

extern "C" void kernel_launch(void* const* d_in, const int* in_sizes, int n_in,
                              void* d_out, int out_size, void* d_ws, size_t ws_size,
                              hipStream_t stream) {
    const float* x    = (const float*)d_in[0];
    const float* nw   = (const float*)d_in[1];
    const float* nb   = (const float*)d_in[2];
    const float* qkvw = (const float*)d_in[3];
    const float* qkvb = (const float*)d_in[4];
    const float* pw   = (const float*)d_in[5];
    const float* pb   = (const float*)d_in[6];
    float* out = (float*)d_out;

    char* ws = (char*)d_ws;
    const size_t SZ_ACT = (size_t)NROWS * DHEAD * sizeof(_Float16);   // 4 MB
    _Float16* xn      = (_Float16*)(ws);
    _Float16* qkvw_h  = (_Float16*)(ws + SZ_ACT);                     // 384 KB
    _Float16* projw_h = (_Float16*)(ws + SZ_ACT + 393216);            // 128 KB
    _Float16* qh      = (_Float16*)(ws + SZ_ACT + 524288);
    _Float16* kh      = (_Float16*)(ws + 2 * SZ_ACT + 524288);
    _Float16* vh      = (_Float16*)(ws + 3 * SZ_ACT + 524288);
    _Float16* ao      = (_Float16*)(ws + 4 * SZ_ACT + 524288);

    ln_kernel  <<<NROWS, 256, 0, stream>>>(x, nw, nb, xn);
    cvt_kernel <<<768,   256, 0, stream>>>(qkvw, pw, qkvw_h, projw_h);
    qkv_gemm   <<<6144,  128, 0, stream>>>(xn, qkvw_h, qkvb, qh, kh, vh);   // 512*48 tiles
    flash_kernel<<<128,  128, 0, stream>>>(qh, kh, vh, ao);
    proj_gemm  <<<2048,  128, 0, stream>>>(ao, projw_h, pb, x, out);        // 512*16 tiles
}
